// DeformableAttnDecoderLayer_24429773980429
// MI455X (gfx1250) — matrix operand order repaired and checked by reading, not compiled, not run on hardware
//
#include <hip/hip_runtime.h>
#include <hip/hip_bf16.h>
#include <math.h>

// ---------------------------------------------------------------------------
// Problem constants (from the reference; spatial_shapes/level_start_index are
// compile-time constants, so the int64 inputs 4/5 are ignored).
// ---------------------------------------------------------------------------
#define DMODEL   256
#define DFFN     1024
#define NHEADS   8
#define NLEVELS  4
#define NPOINTS  4
#define DHEAD    32
#define BATCH    2
#define LQn      8192
#define SRC_LEN  21760            // 128*128 + 64*64 + 32*32 + 16*16
#define MQ       (BATCH * LQn)    // 16384 query rows
#define MV       (BATCH * SRC_LEN)// 43520 value rows

typedef __bf16 bf16_t;
typedef __attribute__((ext_vector_type(16))) __bf16 bf16x16;
typedef __attribute__((ext_vector_type(8)))  __bf16 bf16x8;
typedef __attribute__((ext_vector_type(8)))  float  f32x8;

union AFrag { bf16x16 v; bf16x8 h[2]; };

#define FLAG_RELU    1
#define FLAG_BF16OUT 2

// ---------------------------------------------------------------------------
// Elementwise prep kernels
// ---------------------------------------------------------------------------
__global__ void __launch_bounds__(256)
f32_to_bf16_kernel(const float* __restrict__ a, bf16_t* __restrict__ o, int n) {
    int i = blockIdx.x * blockDim.x + threadIdx.x;
    if (i < n) o[i] = (bf16_t)a[i];
}

__global__ void __launch_bounds__(256)
add_to_bf16_kernel(const float* __restrict__ a, const float* __restrict__ b,
                   bf16_t* __restrict__ o, int n) {
    int i = blockIdx.x * blockDim.x + threadIdx.x;
    if (i < n) o[i] = (bf16_t)(a[i] + b[i]);
}

// W[K][N] (row-major f32) -> Wt[N][K] (row-major bf16) so the WMMA B-fragment
// becomes a contiguous per-lane 32B load.
__global__ void __launch_bounds__(256)
transpose_to_bf16_kernel(const float* __restrict__ W, bf16_t* __restrict__ Wt,
                         int K, int N) {
    int i = blockIdx.x * blockDim.x + threadIdx.x;
    if (i >= K * N) return;
    int k = i / N, n = i - k * N;
    Wt[(size_t)n * K + k] = (bf16_t)W[i];
}

// ---------------------------------------------------------------------------
// WMMA GEMM: C[M][N] = A[M][K](bf16, row-major) * Wt[N][K](bf16, transposed)
//            + bias[N]; optional ReLU; f32 or bf16 output.
//
// One wave32 per 16x64 C strip (4 accumulators): one A fragment is amortized
// over 4 WMMAs.  The K loop is a ping-pong over two fragment register sets,
// two 32-wide slabs per iteration, so the double buffer rotates structurally
// (no v_mov swap copies) and loads for slab i+1 are in flight during slab i's
// WMMAs.  K is a template constant (256 / 1024, both % 64 == 0) so the loop
// fully unrolls; N % 64 == 0 (128/256/1024 here).
//
// Fragment layouts used here (A and B share contiguous K order, not the ISA table order):
//   A (16x32 bf16): lane L -> row M=L&15, half=L>>4; 16 contiguous K at
//       k0+16*half (matches B)
//   B (32x16 bf16): lane L -> col N=L&15, half=L>>4; 16 contiguous K at
//       k0+16*half
//   C (16x16 f32): lane L -> col N=L&15; acc[r] -> row M = r + 8*(L>>4)
// ---------------------------------------------------------------------------
template<int K>
__global__ void __launch_bounds__(256)
wmma_gemm_bf16_kernel(const bf16_t* __restrict__ A, const bf16_t* __restrict__ Wt,
                      const float* __restrict__ bias, void* __restrict__ Cout,
                      int M, int N, int flags) {
    constexpr int NT = 4;                       // 4 x 16-wide N tiles per wave
    static_assert(K % 64 == 0, "K must be a multiple of 64");
    const int wave = blockIdx.x * (blockDim.x >> 5) + (threadIdx.x >> 5);
    const int lane = threadIdx.x & 31;
    const int groupsN = N >> 6;                 // 64-wide N groups
    const int tilesM  = M >> 4;
    if (wave >= tilesM * groupsN) return;
    const int tileM = (wave / groupsN) << 4;
    const int tileN = (wave % groupsN) << 6;

    const int mn   = lane & 15;   // A row within tile == B/C column within tile
    const int half = lane >> 4;

    const bf16_t* aPtr = A  + (size_t)(tileM + mn) * K + 16 * half;
    const bf16_t* bPtr = Wt + (size_t)(tileN + mn) * K + 16 * half;
    const size_t  bStride = (size_t)16 * K;     // next 16-wide N tile of Wt

    f32x8 acc[NT] = {};

    AFrag   fa0, fa1;
    bf16x16 fb0[NT], fb1[NT];

    // prologue: slab 0 -> buffer 0
    fa0.h[0] = *(const bf16x8*)(aPtr);
    fa0.h[1] = *(const bf16x8*)(aPtr + 8);
    #pragma unroll
    for (int t = 0; t < NT; ++t)
        fb0[t] = *(const bf16x16*)(bPtr + t * bStride);

    #pragma unroll
    for (int k0 = 0; k0 < K; k0 += 64) {
        // slab (k0+32) -> buffer 1 (always exists since K % 64 == 0)
        {
            const bf16_t* aN = aPtr + k0 + 32;
            const bf16_t* bN = bPtr + k0 + 32;
            fa1.h[0] = *(const bf16x8*)(aN);
            fa1.h[1] = *(const bf16x8*)(aN + 8);
            #pragma unroll
            for (int t = 0; t < NT; ++t)
                fb1[t] = *(const bf16x16*)(bN + t * bStride);
        }
        if (k0 + 96 < K) {  // gfx1250 global_prefetch_b8, two slabs ahead
            __builtin_prefetch(aPtr + k0 + 96, 0, 1);
            __builtin_prefetch(bPtr + k0 + 96, 0, 1);
        }
        #pragma unroll
        for (int t = 0; t < NT; ++t)
            acc[t] = __builtin_amdgcn_wmma_f32_16x16x32_bf16(
                false, fa0.v, false, fb0[t], (short)0, acc[t], false, false);
        // slab (k0+64) -> buffer 0 (skipped statically on the last iteration)
        if (k0 + 64 < K) {
            const bf16_t* aN = aPtr + k0 + 64;
            const bf16_t* bN = bPtr + k0 + 64;
            fa0.h[0] = *(const bf16x8*)(aN);
            fa0.h[1] = *(const bf16x8*)(aN + 8);
            #pragma unroll
            for (int t = 0; t < NT; ++t)
                fb0[t] = *(const bf16x16*)(bN + t * bStride);
        }
        #pragma unroll
        for (int t = 0; t < NT; ++t)
            acc[t] = __builtin_amdgcn_wmma_f32_16x16x32_bf16(
                false, fa1.v, false, fb1[t], (short)0, acc[t], false, false);
    }

    #pragma unroll
    for (int t = 0; t < NT; ++t) {
        const int colBase = tileN + 16 * t + mn;
        const float bv = bias ? bias[colBase] : 0.0f;
        if (flags & FLAG_BF16OUT) {
            bf16_t* C = (bf16_t*)Cout;
            #pragma unroll
            for (int r = 0; r < 8; ++r) {
                float v = acc[t][r] + bv;
                if (flags & FLAG_RELU) v = fmaxf(v, 0.0f);
                C[(size_t)(tileM + r + 8 * half) * N + colBase] = (bf16_t)v;
            }
        } else {
            float* C = (float*)Cout;
            #pragma unroll
            for (int r = 0; r < 8; ++r) {
                float v = acc[t][r] + bv;
                if (flags & FLAG_RELU) v = fmaxf(v, 0.0f);
                C[(size_t)(tileM + r + 8 * half) * N + colBase] = v;
            }
        }
    }
}

// ---------------------------------------------------------------------------
// Deformable sampling: one wave per (b, q, head); lane = channel d (DHEAD=32
// exactly fills a wave32 -> coalesced b32 gathers). Softmax over the 16
// (level, point) logits is computed redundantly per lane in f32.
// ---------------------------------------------------------------------------
__global__ void __launch_bounds__(256)
ms_deform_sample_kernel(const float* __restrict__ value,   // [B][S][256] (c = h*32+d)
                        const float* __restrict__ off,     // [B*LQ][256]
                        const float* __restrict__ attnlog, // [B*LQ][128]
                        const float* __restrict__ refpts,  // [B*LQ][4][2]
                        bf16_t* __restrict__ out)          // [B*LQ][256] bf16
{
    const int wave = blockIdx.x * (blockDim.x >> 5) + (threadIdx.x >> 5);
    const int lane = threadIdx.x & 31;
    const int h  = wave & (NHEADS - 1);
    const int bq = wave >> 3;
    if (bq >= MQ) return;
    const int b = bq >> 13;  // bq / LQ

    const int Hs[NLEVELS]     = {128, 64, 32, 16};
    const int Ws[NLEVELS]     = {128, 64, 32, 16};
    const int starts[NLEVELS] = {0, 16384, 20480, 21504};

    // softmax over the 16 (level, point) logits for this head
    const float* lg = attnlog + (size_t)bq * (NHEADS * NLEVELS * NPOINTS) + h * 16;
    float w[16], mx = -3.4e38f;
    #pragma unroll
    for (int j = 0; j < 16; ++j) { w[j] = lg[j]; mx = fmaxf(mx, w[j]); }
    float s = 0.0f;
    #pragma unroll
    for (int j = 0; j < 16; ++j) { w[j] = __expf(w[j] - mx); s += w[j]; }
    const float inv = 1.0f / s;

    const float* offp = off + (size_t)bq * 256 + h * (NLEVELS * NPOINTS * 2);
    const float* refp = refpts + (size_t)bq * (NLEVELS * 2);

    float acc = 0.0f;
    #pragma unroll
    for (int l = 0; l < NLEVELS; ++l) {
        const int Hl = Hs[l], Wl = Ws[l];
        const float rx = refp[2 * l], ry = refp[2 * l + 1];
        const float* vlev = value + ((size_t)b * SRC_LEN + starts[l]) * DMODEL
                                  + h * DHEAD + lane;
        #pragma unroll
        for (int p = 0; p < NPOINTS; ++p) {
            const float ox = offp[(l * NPOINTS + p) * 2 + 0];
            const float oy = offp[(l * NPOINTS + p) * 2 + 1];
            const float x = rx * (float)Wl + ox - 0.5f;   // (ref + off/W)*W - 0.5
            const float y = ry * (float)Hl + oy - 0.5f;
            const float x0f = floorf(x), y0f = floorf(y);
            const int   x0 = (int)x0f,  y0 = (int)y0f;
            const float wx = x - x0f,   wy = y - y0f;
            const float aw = w[l * NPOINTS + p] * inv;
            const float s00 = (1.0f - wx) * (1.0f - wy);
            const float s10 = wx * (1.0f - wy);
            const float s01 = (1.0f - wx) * wy;
            const float s11 = wx * wy;
            auto corner = [&](int xi, int yi, float sc) {
                if (xi >= 0 && xi < Wl && yi >= 0 && yi < Hl)
                    acc += aw * sc * vlev[(size_t)(yi * Wl + xi) * DMODEL];
            };
            corner(x0,     y0,     s00);
            corner(x0 + 1, y0,     s10);
            corner(x0,     y0 + 1, s01);
            corner(x0 + 1, y0 + 1, s11);
        }
    }
    out[(size_t)bq * DMODEL + h * DHEAD + lane] = (bf16_t)acc;
}

// ---------------------------------------------------------------------------
// y = LayerNorm(a + b) * g + beta ; one wave per 256-wide row, lane owns 8
// contiguous channels (2x float4 loads), wave32 shfl_xor reduction.
// Optionally also emits a bf16 copy (feeds the next WMMA GEMM).
// ---------------------------------------------------------------------------
__global__ void __launch_bounds__(256)
add_ln_kernel(const float* __restrict__ a, const float* __restrict__ b,
              const float* __restrict__ g, const float* __restrict__ be,
              float* __restrict__ outf, bf16_t* __restrict__ outb, int rows) {
    const int wave = blockIdx.x * (blockDim.x >> 5) + (threadIdx.x >> 5);
    const int lane = threadIdx.x & 31;
    if (wave >= rows) return;
    const size_t base = (size_t)wave * DMODEL + lane * 8;

    float v[8];
    const float4* a4 = (const float4*)(a + base);
    const float4* b4 = (const float4*)(b + base);
    #pragma unroll
    for (int i = 0; i < 2; ++i) {
        float4 x = a4[i], y = b4[i];
        v[4 * i + 0] = x.x + y.x; v[4 * i + 1] = x.y + y.y;
        v[4 * i + 2] = x.z + y.z; v[4 * i + 3] = x.w + y.w;
    }
    float s = 0.0f, s2 = 0.0f;
    #pragma unroll
    for (int i = 0; i < 8; ++i) { s += v[i]; s2 += v[i] * v[i]; }
    #pragma unroll
    for (int o = 16; o >= 1; o >>= 1) {
        s  += __shfl_xor(s,  o, 32);
        s2 += __shfl_xor(s2, o, 32);
    }
    const float mu  = s * (1.0f / DMODEL);
    const float var = s2 * (1.0f / DMODEL) - mu * mu;
    const float r   = rsqrtf(var + 1e-5f);
    #pragma unroll
    for (int i = 0; i < 8; ++i) {
        const int c = lane * 8 + i;
        const float o = (v[i] - mu) * r * g[c] + be[c];
        outf[base + i] = o;
        if (outb) outb[base + i] = (bf16_t)o;
    }
}

// ---------------------------------------------------------------------------
// Host-side orchestration
// ---------------------------------------------------------------------------
extern "C" void kernel_launch(void* const* d_in, const int* in_sizes, int n_in,
                              void* d_out, int out_size, void* d_ws, size_t ws_size,
                              hipStream_t stream) {
    (void)in_sizes; (void)n_in; (void)out_size; (void)ws_size;

    const float* tgt       = (const float*)d_in[0];
    const float* query_pos = (const float*)d_in[1];
    const float* refpts    = (const float*)d_in[2];
    const float* src       = (const float*)d_in[3];
    // d_in[4] spatial_shapes, d_in[5] level_start_index: compile-time constants
    const float* W_value   = (const float*)d_in[6];
    const float* b_value   = (const float*)d_in[7];
    const float* W_off     = (const float*)d_in[8];
    const float* b_off     = (const float*)d_in[9];
    const float* W_attn    = (const float*)d_in[10];
    const float* b_attn    = (const float*)d_in[11];
    const float* W_out     = (const float*)d_in[12];
    const float* b_out     = (const float*)d_in[13];
    const float* ln1_g     = (const float*)d_in[14];
    const float* ln1_b     = (const float*)d_in[15];
    const float* W1        = (const float*)d_in[16];
    const float* b1        = (const float*)d_in[17];
    const float* W2        = (const float*)d_in[18];
    const float* b2        = (const float*)d_in[19];
    const float* ln3_g     = (const float*)d_in[20];
    const float* ln3_b     = (const float*)d_in[21];

    // ---- workspace layout (~193 MB) ----
    char* ws = (char*)d_ws;
    size_t o = 0;
    auto alloc = [&](size_t bytes) {
        void* p = ws + o;
        o += (bytes + 255) & ~(size_t)255;
        return p;
    };
    bf16_t* q_bf     = (bf16_t*)alloc((size_t)MQ * DMODEL * 2);
    bf16_t* src_bf   = (bf16_t*)alloc((size_t)MV * DMODEL * 2);
    float*  value    = (float*) alloc((size_t)MV * DMODEL * 4);
    float*  off_f    = (float*) alloc((size_t)MQ * 256 * 4);
    float*  attnlog  = (float*) alloc((size_t)MQ * 128 * 4);
    bf16_t* attn_bf  = (bf16_t*)alloc((size_t)MQ * DMODEL * 2);
    float*  tgt2     = (float*) alloc((size_t)MQ * DMODEL * 4);
    float*  x_f      = (float*) alloc((size_t)MQ * DMODEL * 4);
    bf16_t* x_bf     = (bf16_t*)alloc((size_t)MQ * DMODEL * 2);
    bf16_t* hid_bf   = (bf16_t*)alloc((size_t)MQ * DFFN * 2);
    float*  ff_f     = (float*) alloc((size_t)MQ * DMODEL * 4);
    bf16_t* Wv_t     = (bf16_t*)alloc((size_t)256 * 256 * 2);
    bf16_t* Woff_t   = (bf16_t*)alloc((size_t)256 * 256 * 2);
    bf16_t* Wattn_t  = (bf16_t*)alloc((size_t)256 * 128 * 2);
    bf16_t* Wout_t   = (bf16_t*)alloc((size_t)256 * 256 * 2);
    bf16_t* W1_t     = (bf16_t*)alloc((size_t)256 * 1024 * 2);
    bf16_t* W2_t     = (bf16_t*)alloc((size_t)1024 * 256 * 2);

    const int T = 256;
    auto cdiv = [](long long a, long long b) { return (int)((a + b - 1) / b); };

    // ---- weight transposes (f32 -> bf16, [K][N] -> [N][K]) ----
    transpose_to_bf16_kernel<<<cdiv(256 * 256, T), T, 0, stream>>>(W_value, Wv_t, 256, 256);
    transpose_to_bf16_kernel<<<cdiv(256 * 256, T), T, 0, stream>>>(W_off, Woff_t, 256, 256);
    transpose_to_bf16_kernel<<<cdiv(256 * 128, T), T, 0, stream>>>(W_attn, Wattn_t, 256, 128);
    transpose_to_bf16_kernel<<<cdiv(256 * 256, T), T, 0, stream>>>(W_out, Wout_t, 256, 256);
    transpose_to_bf16_kernel<<<cdiv(256 * 1024, T), T, 0, stream>>>(W1, W1_t, 256, 1024);
    transpose_to_bf16_kernel<<<cdiv(1024 * 256, T), T, 0, stream>>>(W2, W2_t, 1024, 256);

    // ---- activations to bf16 ----
    f32_to_bf16_kernel<<<cdiv((long long)MV * DMODEL, T), T, 0, stream>>>(src, src_bf, MV * DMODEL);
    add_to_bf16_kernel<<<cdiv((long long)MQ * DMODEL, T), T, 0, stream>>>(tgt, query_pos, q_bf, MQ * DMODEL);

    // ---- projections on the WMMA pipe (one wave per 16x64 C strip) ----
    auto gemm_blocks = [&](int M, int N) { return cdiv((long long)(M >> 4) * (N >> 6), 8); };
    wmma_gemm_bf16_kernel<256><<<gemm_blocks(MV, 256), T, 0, stream>>>(src_bf, Wv_t, b_value, value, MV, 256, 0);
    wmma_gemm_bf16_kernel<256><<<gemm_blocks(MQ, 256), T, 0, stream>>>(q_bf, Woff_t, b_off, off_f, MQ, 256, 0);
    wmma_gemm_bf16_kernel<256><<<gemm_blocks(MQ, 128), T, 0, stream>>>(q_bf, Wattn_t, b_attn, attnlog, MQ, 128, 0);

    // ---- softmax + bilinear deformable sampling ----
    ms_deform_sample_kernel<<<cdiv((long long)MQ * NHEADS, 8), T, 0, stream>>>(
        value, off_f, attnlog, refpts, attn_bf);

    // ---- output projection + residual/LN1 ----
    wmma_gemm_bf16_kernel<256><<<gemm_blocks(MQ, 256), T, 0, stream>>>(attn_bf, Wout_t, b_out, tgt2, MQ, 256, 0);
    add_ln_kernel<<<cdiv(MQ, 8), T, 0, stream>>>(tgt, tgt2, ln1_g, ln1_b, x_f, x_bf, MQ);

    // ---- FFN (fused ReLU into GEMM1, bf16 hidden) + residual/LN3 -> d_out ----
    wmma_gemm_bf16_kernel<256><<<gemm_blocks(MQ, 1024), T, 0, stream>>>(
        x_bf, W1_t, b1, hid_bf, MQ, 1024, FLAG_RELU | FLAG_BF16OUT);
    wmma_gemm_bf16_kernel<1024><<<gemm_blocks(MQ, 256), T, 0, stream>>>(
        hid_bf, W2_t, b2, ff_f, MQ, 256, 0);
    add_ln_kernel<<<cdiv(MQ, 8), T, 0, stream>>>(x_f, ff_f, ln3_g, ln3_b, (float*)d_out, (bf16_t*)nullptr, MQ);
}
